// FraudGNN_68513318306179
// MI455X (gfx1250) — matrix operand order repaired and checked by reading, not compiled, not run on hardware
//
#include <hip/hip_runtime.h>

typedef _Float16 f16;
typedef __attribute__((ext_vector_type(16))) _Float16 v16h;
typedef __attribute__((ext_vector_type(8)))  float    v8f;

// ---------- helpers ----------
__device__ __forceinline__ unsigned f2ord(float f) {
    unsigned u = __float_as_uint(f);
    return (u & 0x80000000u) ? ~u : (u | 0x80000000u);   // order-preserving float->uint
}
__device__ __forceinline__ float ord2f(unsigned u) {
    return __uint_as_float((u & 0x80000000u) ? (u & 0x7fffffffu) : ~u);
}
__device__ __forceinline__ float lrelu02(float v) { return v > 0.f ? v : 0.2f * v; }
__device__ __forceinline__ float elu1(float v)    { return v > 0.f ? v : (__expf(v) - 1.f); }

// ============================================================================
// Fragment packing. WMMA 16x16x32 f16 per-lane layouts (ISA 7.12.2):
//  A (16x32): lane = lh*16+mr holds row mr, K = {lh*8+e} e<8, {16+lh*8+(e-8)} e>=8
//  B (32x16): lane = lh*16+nr holds col nr, K = {lh*8+e} e<8, {16+lh*8+(e-8)} e>=8
// Packed buffers: tiles of 512 f16 ( [lane][e] ), A tiles ordered [rowtile][ktile],
// B tiles ordered [ktile][coltile].  Each lane's fragment = 32 contiguous bytes.
// ============================================================================
__global__ void pack_a_f16(const float* __restrict__ src, f16* __restrict__ dst,
                           int M, int K) {
    int t = blockIdx.x * blockDim.x + threadIdx.x;
    if (t >= M * K) return;
    int tile = t >> 9, le = t & 511;
    int lane = le >> 4, e = le & 15;
    int Kt = K >> 5;
    int rt = tile / Kt, kt = tile - rt * Kt;
    int mr = lane & 15, lh = lane >> 4;
    int kk = (e < 8) ? (lh * 8 + e) : (16 + lh * 8 + (e - 8));
    dst[t] = (f16)src[(size_t)(rt * 16 + mr) * K + kt * 32 + kk];
}

__global__ void pack_a_bias_elu_f16(const float* __restrict__ src,
                                    const float* __restrict__ bias,
                                    f16* __restrict__ dst, int M, int K) {
    int t = blockIdx.x * blockDim.x + threadIdx.x;
    if (t >= M * K) return;
    int tile = t >> 9, le = t & 511;
    int lane = le >> 4, e = le & 15;
    int Kt = K >> 5;
    int rt = tile / Kt, kt = tile - rt * Kt;
    int mr = lane & 15, lh = lane >> 4;
    int kk = (e < 8) ? (lh * 8 + e) : (16 + lh * 8 + (e - 8));
    int k = kt * 32 + kk;
    dst[t] = (f16)elu1(src[(size_t)(rt * 16 + mr) * K + k] + bias[k]);
}

__global__ void pack_b_f16(const float* __restrict__ W, f16* __restrict__ dst,
                           int K, int Nc) {
    int t = blockIdx.x * blockDim.x + threadIdx.x;
    if (t >= K * Nc) return;
    int tile = t >> 9, le = t & 511;
    int lane = le >> 4, e = le & 15;
    int Nct = Nc >> 4;
    int kt = tile / Nct, ct = tile - kt * Nct;
    int lh = lane >> 4, nr = lane & 15;
    int kk = (e < 8) ? (lh * 8 + e) : (16 + lh * 8 + (e - 8));
    dst[t] = (f16)W[(size_t)(kt * 32 + kk) * Nc + ct * 16 + nr];
}

// ============================================================================
// GEMM: C[M, NCT*16] = A[M, KT*32] * B ; pre-packed f16 fragments, f32 out.
// gridDim.x = M/16 ; blockDim.x = NCT*32. A strip (KT KB) staged via
// global_load_async_to_lds_b128 (wave 0), fragments re-read as ds_load_b128.
// ============================================================================
template <int KT, int NCT>
__global__ void gemm_wmma_packed(const f16* __restrict__ Ap, const f16* __restrict__ Bp,
                                 float* __restrict__ C) {
    __shared__ f16 sA[KT * 512];
    const int lane = threadIdx.x & 31;
    const int wave = threadIdx.x >> 5;

    if (wave == 0) {
        const char* gsrc = (const char*)(Ap + (size_t)blockIdx.x * KT * 512);
        unsigned lbase = (unsigned)(size_t)(&sA[0]);        // low 32b = LDS offset
        unsigned long long gbase = (unsigned long long)(size_t)gsrc;
#pragma unroll
        for (int i = 0; i < KT * 2; ++i) {                  // 512 B per issue
            unsigned laddr = lbase + i * 512 + lane * 16;
            unsigned long long gaddr = gbase + i * 512 + lane * 16;
            asm volatile("global_load_async_to_lds_b128 %0, %1, off"
                         :: "v"(laddr), "v"(gaddr) : "memory");
        }
        asm volatile("s_wait_asynccnt 0x0" ::: "memory");
    }
    __syncthreads();

    const v16h* sAv = (const v16h*)sA;
    v8f acc = {};
#pragma unroll
    for (int kt = 0; kt < KT; ++kt) {
        v16h a = sAv[kt * 32 + lane];
        v16h b = *(const v16h*)(Bp + (size_t)(kt * NCT + wave) * 512 + lane * 16);
        acc = __builtin_amdgcn_wmma_f32_16x16x32_f16(false, a, false, b,
                                                     (short)0, acc, false, false);
    }

    // C/D layout: VGPR r -> row = r + 8*lane_half, col = lane&15
    const int lh = lane >> 4, mr = lane & 15;
    const int Nc = NCT * 16;
    float* Cp = C + (size_t)(blockIdx.x * 16 + lh * 8) * Nc + wave * 16 + mr;
#pragma unroll
    for (int r = 0; r < 8; ++r) Cp[(size_t)r * Nc] = acc[r];
}

// ---------- attention dot products: one wave per (node, head), C=64 ----------
__global__ void att_dots(const float* __restrict__ h, const float* __restrict__ a_src,
                         const float* __restrict__ a_dst, float* __restrict__ as_,
                         float* __restrict__ ad_, int n_nodes, int H) {
    int gw   = (blockIdx.x * blockDim.x + threadIdx.x) >> 5;
    int lane = threadIdx.x & 31;
    if (gw >= n_nodes * H) return;
    int node = gw / H, head = gw - node * H;
    const float* hp  = h + (size_t)node * H * 64 + head * 64;
    const float* asp = a_src + head * 64;
    const float* adp = a_dst + head * 64;
    float s = hp[lane] * asp[lane] + hp[lane + 32] * asp[lane + 32];
    float d = hp[lane] * adp[lane] + hp[lane + 32] * adp[lane + 32];
#pragma unroll
    for (int off = 16; off > 0; off >>= 1) {
        s += __shfl_down(s, off, 32);
        d += __shfl_down(d, off, 32);
    }
    if (lane == 0) { as_[gw] = s; ad_[gw] = d; }
}

// ---------- segment softmax pass 1: atomic max of logits ----------
__global__ void edge_max(const int* __restrict__ ei, int E, int n_nodes, int H,
                         const float* __restrict__ as_, const float* __restrict__ ad_,
                         unsigned* __restrict__ menc) {
    int t = blockIdx.x * blockDim.x + threadIdx.x;
    if (t >= (E + n_nodes) * H) return;
    int e = t / H, head = t - e * H;
    int src, dst;
    if (e < E) { src = ei[e]; dst = ei[E + e]; } else { src = dst = e - E; }
    float l = lrelu02(as_[src * H + head] + ad_[dst * H + head]);
    atomicMax(&menc[dst * H + head], f2ord(l));
}

// ---------- pass 2: ex = exp(l - m); segment sum ----------
__global__ void edge_exp(const int* __restrict__ ei, int E, int n_nodes, int H,
                         const float* __restrict__ as_, const float* __restrict__ ad_,
                         const unsigned* __restrict__ menc, float* __restrict__ ex,
                         float* __restrict__ ssum) {
    int t = blockIdx.x * blockDim.x + threadIdx.x;
    if (t >= (E + n_nodes) * H) return;
    int e = t / H, head = t - e * H;
    int src, dst;
    if (e < E) { src = ei[e]; dst = ei[E + e]; } else { src = dst = e - E; }
    float l = lrelu02(as_[src * H + head] + ad_[dst * H + head]);
    float v = __expf(l - ord2f(menc[dst * H + head]));
    ex[t] = v;
    atomicAdd(&ssum[dst * H + head], v);
}

// ---------- pass 3: out[dst] += h[src] * alpha ; one wave per edge ----------
__global__ void edge_aggr(const int* __restrict__ ei, int E, int n_nodes, int H,
                          const float* __restrict__ hfeat, const float* __restrict__ ex,
                          const float* __restrict__ ssum, float* __restrict__ outacc) {
    int gw   = (blockIdx.x * blockDim.x + threadIdx.x) >> 5;
    int lane = threadIdx.x & 31;
    if (gw >= E + n_nodes) return;
    int src, dst;
    if (gw < E) { src = ei[gw]; dst = ei[E + gw]; } else { src = dst = gw - E; }
    const int C = H * 64;
    for (int head = 0; head < H; ++head) {
        float alpha = ex[gw * H + head] / (ssum[dst * H + head] + 1e-16f);
        const float* hp = hfeat + (size_t)src * C + head * 64;
        float* op       = outacc + (size_t)dst * C + head * 64;
        atomicAdd(&op[lane],      hp[lane]      * alpha);
        atomicAdd(&op[lane + 32], hp[lane + 32] * alpha);
    }
}

// ---------- bias + elu, f32 out ----------
__global__ void bias_elu_f32(const float* __restrict__ acc, const float* __restrict__ bias,
                             float* __restrict__ out, int n, int C) {
    int i = blockIdx.x * blockDim.x + threadIdx.x;
    if (i >= n) return;
    out[i] = elu1(acc[i] + bias[i % C]);
}

// ---------- classifier: relu(h@W3+b3)@W4+b4 ; one thread per node ----------
__global__ void classifier_kernel(const float* __restrict__ h,
                                  const float* __restrict__ W3, const float* __restrict__ b3,
                                  const float* __restrict__ W4, const float* __restrict__ b4,
                                  float* __restrict__ out, int n_nodes) {
    __shared__ float sW3[64 * 32];
    __shared__ float sW4[64];
    for (int i = threadIdx.x; i < 64 * 32; i += blockDim.x) sW3[i] = W3[i];
    for (int i = threadIdx.x; i < 64; i += blockDim.x) sW4[i] = W4[i];
    __syncthreads();
    int nid = blockIdx.x * blockDim.x + threadIdx.x;
    if (nid >= n_nodes) return;
    const float* hp = h + (size_t)nid * 64;
    float hl[64];
#pragma unroll
    for (int k = 0; k < 64; ++k) hl[k] = hp[k];
    float o0 = b4[0], o1 = b4[1];
    for (int j = 0; j < 32; ++j) {
        float s = b3[j];
#pragma unroll
        for (int k = 0; k < 64; ++k) s += hl[k] * sW3[k * 32 + j];
        s = s > 0.f ? s : 0.f;
        o0 += s * sW4[j * 2 + 0];
        o1 += s * sW4[j * 2 + 1];
    }
    out[(size_t)nid * 2 + 0] = o0;
    out[(size_t)nid * 2 + 1] = o1;
}

extern "C" void kernel_launch(void* const* d_in, const int* in_sizes, int n_in,
                              void* d_out, int out_size, void* d_ws, size_t ws_size,
                              hipStream_t stream) {
    const float* x      = (const float*)d_in[0];
    const int*   ei     = (const int*)d_in[1];
    const float* W1     = (const float*)d_in[2];
    const float* a_src1 = (const float*)d_in[3];
    const float* a_dst1 = (const float*)d_in[4];
    const float* b1     = (const float*)d_in[5];
    const float* W2     = (const float*)d_in[6];
    const float* a_src2 = (const float*)d_in[7];
    const float* a_dst2 = (const float*)d_in[8];
    const float* b2     = (const float*)d_in[9];
    const float* W3     = (const float*)d_in[10];
    const float* b3     = (const float*)d_in[11];
    const float* W4     = (const float*)d_in[12];
    const float* b4     = (const float*)d_in[13];
    float* out = (float*)d_out;

    const int N   = in_sizes[0] / 128;   // 50000
    const int E   = in_sizes[1] / 2;     // 800000
    const int TOT = E + N;

    // ---- workspace layout ----
    char*  ws  = (char*)d_ws;
    size_t off = 0;
    auto alloc = [&](size_t bytes) -> char* {
        char* p = ws + off;
        off += (bytes + 255) & ~(size_t)255;
        return p;
    };
    f16*      w1p  = (f16*)alloc((size_t)128 * 256 * sizeof(f16));
    f16*      w2p  = (f16*)alloc((size_t)256 * 64 * sizeof(f16));
    float*    h1   = (float*)alloc((size_t)N * 256 * sizeof(float)); // layer-2 aliases here
    float*    out1 = (float*)alloc((size_t)N * 256 * sizeof(float));
    f16*      h1ep = (f16*)alloc((size_t)N * 256 * sizeof(f16));     // packed layer-2 A
    f16*      xp   = (f16*)alloc((size_t)N * 128 * sizeof(f16));     // packed layer-1 A
    float*    as1  = (float*)alloc((size_t)N * 4 * sizeof(float));
    float*    ad1  = (float*)alloc((size_t)N * 4 * sizeof(float));
    unsigned* m1   = (unsigned*)alloc((size_t)N * 4 * sizeof(unsigned));
    float*    s1   = (float*)alloc((size_t)N * 4 * sizeof(float));
    float*    ex1  = (float*)alloc((size_t)TOT * 4 * sizeof(float));
    // layer-2 buffers alias the (dead after layer-1) h1 region: ~42.6MB <= 51.2MB
    char* l2 = (char*)h1;
    float*    h2   = (float*)l2; l2 += (size_t)N * 64 * sizeof(float);
    float*    out2 = (float*)l2; l2 += (size_t)N * 64 * sizeof(float);
    float*    h2o  = (float*)l2; l2 += (size_t)N * 64 * sizeof(float);
    float*    as2  = (float*)l2; l2 += (size_t)N * sizeof(float);
    float*    ad2  = (float*)l2; l2 += (size_t)N * sizeof(float);
    unsigned* m2   = (unsigned*)l2; l2 += (size_t)N * sizeof(unsigned);
    float*    s2   = (float*)l2; l2 += (size_t)N * sizeof(float);
    float*    ex2  = (float*)l2; l2 += (size_t)TOT * sizeof(float);

    const int BT = 256;
    auto cdiv = [](long long a, long long b) { return (int)((a + b - 1) / b); };

    // ---- pack weights / input into WMMA fragment order ----
    pack_b_f16<<<cdiv(128 * 256, BT), BT, 0, stream>>>(W1, w1p, 128, 256);
    pack_b_f16<<<cdiv(256 * 64, BT), BT, 0, stream>>>(W2, w2p, 256, 64);
    pack_a_f16<<<cdiv((long long)N * 128, BT), BT, 0, stream>>>(x, xp, N, 128);

    // ================= Layer 1 (4 heads x 64) =================
    gemm_wmma_packed<4, 16><<<N / 16, 16 * 32, 0, stream>>>(xp, w1p, h1);
    att_dots<<<cdiv((long long)N * 4 * 32, BT), BT, 0, stream>>>(h1, a_src1, a_dst1, as1, ad1, N, 4);

    hipMemsetAsync(m1, 0, (size_t)N * 4 * sizeof(unsigned), stream);
    hipMemsetAsync(s1, 0, (size_t)N * 4 * sizeof(float), stream);
    hipMemsetAsync(out1, 0, (size_t)N * 256 * sizeof(float), stream);

    edge_max<<<cdiv((long long)TOT * 4, BT), BT, 0, stream>>>(ei, E, N, 4, as1, ad1, m1);
    edge_exp<<<cdiv((long long)TOT * 4, BT), BT, 0, stream>>>(ei, E, N, 4, as1, ad1, m1, ex1, s1);
    edge_aggr<<<cdiv((long long)TOT * 32, BT), BT, 0, stream>>>(ei, E, N, 4, h1, ex1, s1, out1);
    pack_a_bias_elu_f16<<<cdiv((long long)N * 256, BT), BT, 0, stream>>>(out1, b1, h1ep, N, 256);

    // ================= Layer 2 (1 head x 64) =================
    gemm_wmma_packed<8, 4><<<N / 16, 4 * 32, 0, stream>>>(h1ep, w2p, h2);
    att_dots<<<cdiv((long long)N * 32, BT), BT, 0, stream>>>(h2, a_src2, a_dst2, as2, ad2, N, 1);

    hipMemsetAsync(m2, 0, (size_t)N * sizeof(unsigned), stream);
    hipMemsetAsync(s2, 0, (size_t)N * sizeof(float), stream);
    hipMemsetAsync(out2, 0, (size_t)N * 64 * sizeof(float), stream);

    edge_max<<<cdiv(TOT, BT), BT, 0, stream>>>(ei, E, N, 1, as2, ad2, m2);
    edge_exp<<<cdiv(TOT, BT), BT, 0, stream>>>(ei, E, N, 1, as2, ad2, m2, ex2, s2);
    edge_aggr<<<cdiv((long long)TOT * 32, BT), BT, 0, stream>>>(ei, E, N, 1, h2, ex2, s2, out2);
    bias_elu_f32<<<cdiv((long long)N * 64, BT), BT, 0, stream>>>(out2, b2, h2o, N * 64, 64);

    // ================= Classifier =================
    classifier_kernel<<<cdiv(N, BT), BT, 0, stream>>>(h2o, W3, b3, W4, b4, out, N);
    (void)ws_size; (void)n_in; (void)out_size;
}
